// ConformerMultiHeadedSelfAttention_9440338117416
// MI455X (gfx1250) — compile-verified, matrix operand order checked by reading, not run on hardware
//
#include <hip/hip_runtime.h>

// ConformerMultiHeadedSelfAttention (Transformer-XL relative attention), MI455X gfx1250.
// GEMMs via v_wmma_f32_16x16x32_bf16 (wave32); score stage uses TDM tensor_load_to_lds
// to stage 64x64 bf16 tiles into LDS (32KB/block), consumed by 8 waves via ds loads.

typedef __bf16 bf16_t;
typedef __attribute__((ext_vector_type(16))) __bf16 v16bf;
typedef __attribute__((ext_vector_type(8)))  float  v8f;
typedef __attribute__((ext_vector_type(4)))  unsigned int u32x4;
typedef __attribute__((ext_vector_type(4)))  int i32x4;
typedef __attribute__((ext_vector_type(8)))  int i32x8;

static constexpr int B_  = 4;
static constexpr int T_  = 2048;
static constexpr int D_  = 256;
static constexpr int H_  = 4;
static constexpr int DH_ = 64;
static constexpr int BT_ = B_ * T_;          // 8192
static constexpr float SCALE_ = 0.0625f;     // 1/sqrt(256)
static constexpr int WPB_ = 8;               // waves per 256-thread block

__device__ __forceinline__ v8f wmma_bf16(v16bf a, v16bf b, v8f c) {
  return __builtin_amdgcn_wmma_f32_16x16x32_bf16(false, a, false, b, (short)0, c, false, false);
}

// ---- Tensor Data Mover: load a 64x64 bf16 tile (row stride in elements) to LDS.
// D# per CDNA5 ISA ch.8: group0 {count, lds_addr, global_addr, type=2},
// group1 {data_size=2B, tensor dims (large: no OOB clamp), tile 64x64, dim0 stride}.
__device__ __forceinline__ void tdm_load_64x64_bf16(unsigned int lds_off,
                                                    const bf16_t* gptr,
                                                    unsigned int row_stride_elems) {
  const unsigned long long ga = (unsigned long long)(uintptr_t)gptr;
  u32x4 g0;
  g0[0] = 1u;                                           // count=1 (valid), user mode
  g0[1] = lds_off;                                      // lds_addr (bytes)
  g0[2] = (unsigned int)(ga & 0xFFFFFFFFu);             // global_addr[31:0]
  g0[3] = (unsigned int)((ga >> 32) & 0x01FFFFFFu) | (2u << 30);  // addr[56:32] | type=2
  i32x8 g1;
  g1[0] = (int)(1u << 16);          // workgroup_mask=0, data_size=1 (2 bytes)
  g1[1] = 0;                        // atomic_barrier=0, tensor_dim0[15:0]=0 (dim0=1M)
  g1[2] = (int)0x00000010;          // tensor_dim0[31:16]=0x10, tensor_dim1[15:0]=0
  g1[3] = (int)((64u << 16) | 0x10u); // tensor_dim1[31:16]=0x10, tile_dim0=64
  g1[4] = 64;                       // tile_dim1=64, tile_dim2=0
  g1[5] = (int)row_stride_elems;    // tensor_dim0_stride[31:0]
  g1[6] = 0;                        // stride0[47:32]=0, stride1[15:0]=0
  g1[7] = 0;
  i32x4 gz = {};
#if defined(__clang_major__) && (__clang_major__ >= 23)
  i32x8 gz8 = {};
  __builtin_amdgcn_tensor_load_to_lds(g0, g1, gz, gz, gz8, 0);
#else
  __builtin_amdgcn_tensor_load_to_lds(g0, g1, gz, gz, 0);
#endif
}

// ---- fragment loaders (wave32 16x16x32 bf16 layouts, ISA 7.12.2) -------------
__device__ __forceinline__ v16bf frag_a_f32(const float* __restrict__ a, int lda, int lane) {
  const int row = lane & 15;
  const int kb  = (lane >> 4) << 3;
  const float* pr = a + (size_t)row * lda + kb;
  v16bf f;
#pragma unroll
  for (int i = 0; i < 8; ++i) {
    f[i]     = (__bf16)pr[i];
    f[i + 8] = (__bf16)pr[16 + i];
  }
  return f;
}

__device__ __forceinline__ v16bf frag_a_bf16(const bf16_t* a, int lda, int lane) {
  const int row = lane & 15;
  const int kb  = (lane >> 4) << 3;
  const bf16_t* pr = a + (size_t)row * lda + kb;
  v16bf f;
#pragma unroll
  for (int i = 0; i < 8; ++i) {
    f[i]     = pr[i];
    f[i + 8] = pr[16 + i];
  }
  return f;
}

__device__ __forceinline__ v16bf frag_b_f32(const float* __restrict__ b, int ldb, int lane) {
  const int col = lane & 15;
  const int kb  = (lane >> 4) << 3;
  const float* p0 = b + (size_t)kb * ldb + col;
  v16bf f;
#pragma unroll
  for (int i = 0; i < 8; ++i) {
    f[i]     = (__bf16)p0[(size_t)i * ldb];
    f[i + 8] = (__bf16)p0[(size_t)(16 + i) * ldb];
  }
  return f;
}

__device__ __forceinline__ v16bf frag_b_bf16(const bf16_t* __restrict__ b, int ldb, int lane) {
  const int col = lane & 15;
  const int kb  = (lane >> 4) << 3;
  const bf16_t* p0 = b + (size_t)kb * ldb + col;
  v16bf f;
#pragma unroll
  for (int i = 0; i < 8; ++i) {
    f[i]     = p0[(size_t)i * ldb];
    f[i + 8] = p0[(size_t)(16 + i) * ldb];
  }
  return f;
}

// ---- 1) projection GEMM: Y = X(Mx256) @ W(256x256) [+bias], bf16 out --------
__global__ void txl_proj_wmma(const float* __restrict__ X, const float* __restrict__ W,
                              const float* __restrict__ bias,
                              const float* __restrict__ ubias, const float* __restrict__ vbias,
                              bf16_t* __restrict__ out0, bf16_t* __restrict__ out1, int M) {
  const int lane = threadIdx.x & 31;
  const int tile = blockIdx.x * WPB_ + (threadIdx.x >> 5);
  const int colTiles = D_ / 16;
  const int nTiles = (M / 16) * colTiles;
  if (tile >= nTiles) return;
  const int r0 = (tile / colTiles) * 16;
  const int c0 = (tile % colTiles) * 16;
  v8f acc = {};
  for (int kc = 0; kc < D_; kc += 32) {
    v16bf a = frag_a_f32(X + (size_t)r0 * D_ + kc, D_, lane);
    v16bf b = frag_b_f32(W + (size_t)kc * D_ + c0, D_, lane);
    acc = wmma_bf16(a, b, acc);
  }
  const int col   = c0 + (lane & 15);
  const int rbase = r0 + ((lane >> 4) << 3);
  const float bb = bias ? bias[col] : 0.0f;
#pragma unroll
  for (int i = 0; i < 8; ++i) {
    const float v = acc[i] + bb;
    const size_t o = (size_t)(rbase + i) * D_ + col;
    if (out1) {
      out0[o] = (bf16_t)(v + ubias[col]);   // u/v bias flattened (H*DH == D)
      out1[o] = (bf16_t)(v + vbias[col]);
    } else {
      out0[o] = (bf16_t)v;
    }
  }
}

// ---- 2) scores with TDM-staged LDS tiles ------------------------------------
// Block computes a 64x64 macro-tile of content (->d_out attn, raw f32) and
// pos (->bf16 ws). Wave 0 TDM-loads qu/qv/k/p 64x64 bf16 tiles into LDS,
// waits tensorcnt, barrier; 8 waves each compute two 16x16 sub-tiles.
__global__ void txl_score_wmma(const bf16_t* __restrict__ qu, const bf16_t* __restrict__ qv,
                               const bf16_t* __restrict__ kmat, const bf16_t* __restrict__ pmat,
                               float* __restrict__ content, bf16_t* __restrict__ posbuf) {
  __shared__ bf16_t lds_qu[64 * 64];
  __shared__ bf16_t lds_qv[64 * 64];
  __shared__ bf16_t lds_k [64 * 64];
  __shared__ bf16_t lds_p [64 * 64];

  const int lane = threadIdx.x & 31;
  const int wave = threadIdx.x >> 5;
  const int mtT = T_ / 64;                  // 32 macro-tiles per dim
  const int perBH = mtT * mtT;              // 1024
  const int bh  = blockIdx.x / perBH;
  const int rem = blockIdx.x % perBH;
  const int t0 = (rem / mtT) * 64;
  const int s0 = (rem % mtT) * 64;
  const int b = bh / H_, h = bh % H_;

  if (threadIdx.x < 32) {  // one wave issues the async tensor loads
    const size_t qoff = ((size_t)(b * T_ + t0)) * D_ + h * DH_;
    const size_t koff = ((size_t)(b * T_ + s0)) * D_ + h * DH_;
    tdm_load_64x64_bf16((unsigned int)(uintptr_t)&lds_qu[0], qu + qoff, D_);
    tdm_load_64x64_bf16((unsigned int)(uintptr_t)&lds_qv[0], qv + qoff, D_);
    tdm_load_64x64_bf16((unsigned int)(uintptr_t)&lds_k[0],  kmat + koff, D_);
    tdm_load_64x64_bf16((unsigned int)(uintptr_t)&lds_p[0],  pmat + (size_t)s0 * D_ + h * DH_, D_);
    __builtin_amdgcn_s_wait_tensorcnt(0);
  }
  __syncthreads();

#pragma unroll
  for (int sub = 0; sub < 2; ++sub) {
    const int st = wave * 2 + sub;          // 0..15 sub-tiles in 4x4 grid
    const int tl = (st >> 2) * 16;          // local t offset
    const int sl = (st & 3) * 16;           // local s offset
    v8f accC = {}, accP = {};
#pragma unroll
    for (int kc = 0; kc < DH_; kc += 32) {
      // B operand is K^T / P^T: transposed frag of row-major == A pattern
      accC = wmma_bf16(frag_a_bf16(&lds_qu[tl * 64 + kc], 64, lane),
                       frag_a_bf16(&lds_k [sl * 64 + kc], 64, lane), accC);
      accP = wmma_bf16(frag_a_bf16(&lds_qv[tl * 64 + kc], 64, lane),
                       frag_a_bf16(&lds_p [sl * 64 + kc], 64, lane), accP);
    }
    const int scol  = s0 + sl + (lane & 15);
    const int rbase = t0 + tl + ((lane >> 4) << 3);
#pragma unroll
    for (int i = 0; i < 8; ++i) {
      const size_t o = ((size_t)bh * T_ + (rbase + i)) * T_ + scol;
      content[o] = accC[i];
      posbuf[o]  = (bf16_t)accP[i];
    }
  }
}

// ---- 3) relative shift + combine + scale + mask (elementwise, in place) -----
// shift: f=(r+1)*T+c; i=f/(T+1); j=f%(T+1); val = (j==0) ? 0 : pos[i][j-1]
__global__ void txl_shift_mask_scale(float* __restrict__ attn, const bf16_t* __restrict__ posbuf,
                                     const unsigned char* __restrict__ mask) {
  const size_t base = (size_t)blockIdx.x * 2048 + threadIdx.x;
#pragma unroll
  for (int k = 0; k < 8; ++k) {
    const size_t idx = base + (size_t)k * 256;
    const int c = (int)(idx % T_);
    const size_t row = idx / T_;           // == bh*T + r
    const int r  = (int)(row % T_);
    const int bh = (int)(row / T_);
    const int b  = bh / H_;
    const int f  = (r + 1) * T_ + c;
    const int i_ = f / (T_ + 1);
    const int j  = f - i_ * (T_ + 1);
    float sp = 0.0f;
    if (j != 0) sp = (float)posbuf[((size_t)bh * T_ + i_) * T_ + (j - 1)];
    float s = (attn[idx] + sp) * SCALE_;
    if (mask[(size_t)b * T_ + c]) s = -1e9f;
    attn[idx] = s;
  }
}

// ---- 4) row softmax over T=2048, one 256-thread block per row ---------------
__global__ void txl_softmax(float* __restrict__ attn) {
  __shared__ float red[256];
  float* p = attn + (size_t)blockIdx.x * T_;
  const int tid = threadIdx.x;
  float vals[8];
  float m = -3.4e38f;
#pragma unroll
  for (int k = 0; k < 8; ++k) { vals[k] = p[tid + k * 256]; m = fmaxf(m, vals[k]); }
  red[tid] = m; __syncthreads();
  for (int s = 128; s > 0; s >>= 1) {
    if (tid < s) red[tid] = fmaxf(red[tid], red[tid + s]);
    __syncthreads();
  }
  m = red[0]; __syncthreads();
  float sum = 0.0f;
#pragma unroll
  for (int k = 0; k < 8; ++k) { vals[k] = __expf(vals[k] - m); sum += vals[k]; }
  red[tid] = sum; __syncthreads();
  for (int s = 128; s > 0; s >>= 1) {
    if (tid < s) red[tid] += red[tid + s];
    __syncthreads();
  }
  const float inv = 1.0f / red[0];
#pragma unroll
  for (int k = 0; k < 8; ++k) p[tid + k * 256] = vals[k] * inv;
}

// ---- 5) context = attn @ v; one wave computes a full 16x64 output strip -----
// (4 accumulators -> each attn A-fragment is loaded from HBM exactly once)
__global__ void txl_context_wmma(const float* __restrict__ attn, const bf16_t* __restrict__ vmat,
                                 bf16_t* __restrict__ ctx) {
  const int lane = threadIdx.x & 31;
  const int tile = blockIdx.x * WPB_ + (threadIdx.x >> 5);
  const int tT = T_ / 16;                 // 128 row tiles
  if (tile >= B_ * H_ * tT) return;
  const int bh = tile / tT;
  const int t0 = (tile % tT) * 16;
  const int b = bh / H_, h = bh % H_;
  const float*  aB = attn + ((size_t)bh * T_ + t0) * T_;
  const bf16_t* vB = vmat + (size_t)b * T_ * D_ + h * DH_;
  v8f acc[4] = {{}, {}, {}, {}};
  for (int kc = 0; kc < T_; kc += 32) {
    __builtin_prefetch(aB + kc + 128, 0, 1);   // global_prefetch_b8 on streamed attn
    v16bf a = frag_a_f32(aB + kc, T_, lane);
#pragma unroll
    for (int n = 0; n < 4; ++n) {
      v16bf bf = frag_b_bf16(vB + (size_t)kc * D_ + n * 16, D_, lane);
      acc[n] = wmma_bf16(a, bf, acc[n]);
    }
  }
  const int rbase = t0 + ((lane >> 4) << 3);
#pragma unroll
  for (int n = 0; n < 4; ++n) {
    const int col = n * 16 + (lane & 15);
#pragma unroll
    for (int i = 0; i < 8; ++i)
      ctx[((size_t)(b * T_ + rbase + i)) * D_ + h * DH_ + col] = (bf16_t)acc[n][i];
  }
}

// ---- 6) out = ctx(8192x256) @ Wout + bout, f32 out --------------------------
__global__ void txl_outproj_wmma(const bf16_t* __restrict__ ctx, const float* __restrict__ Wout,
                                 const float* __restrict__ bout, float* __restrict__ out) {
  const int lane = threadIdx.x & 31;
  const int tile = blockIdx.x * WPB_ + (threadIdx.x >> 5);
  const int colTiles = D_ / 16;
  if (tile >= (BT_ / 16) * colTiles) return;
  const int r0 = (tile / colTiles) * 16;
  const int c0 = (tile % colTiles) * 16;
  v8f acc = {};
  for (int kc = 0; kc < D_; kc += 32) {
    v16bf a = frag_a_bf16(ctx + (size_t)r0 * D_ + kc, D_, lane);
    v16bf b = frag_b_f32(Wout + (size_t)kc * D_ + c0, D_, lane);
    acc = wmma_bf16(a, b, acc);
  }
  const int col   = c0 + (lane & 15);
  const int rbase = r0 + ((lane >> 4) << 3);
  const float bb = bout[col];
#pragma unroll
  for (int i = 0; i < 8; ++i)
    out[(size_t)(rbase + i) * D_ + col] = acc[i] + bb;
}

extern "C" void kernel_launch(void* const* d_in, const int* in_sizes, int n_in,
                              void* d_out, int out_size, void* d_ws, size_t ws_size,
                              hipStream_t stream) {
  const float* query    = (const float*)d_in[0];
  const float* key      = (const float*)d_in[1];
  const float* value    = (const float*)d_in[2];
  const unsigned char* mask = (const unsigned char*)d_in[3];
  const float* encoding = (const float*)d_in[4];
  const float* Wq   = (const float*)d_in[5];
  const float* bq   = (const float*)d_in[6];
  const float* Wk   = (const float*)d_in[7];
  const float* Wv   = (const float*)d_in[8];
  const float* Wpos = (const float*)d_in[9];
  const float* ub   = (const float*)d_in[10];
  const float* vb   = (const float*)d_in[11];
  const float* Wout = (const float*)d_in[12];
  const float* bout = (const float*)d_in[13];

  float* out_ptr  = (float*)d_out;                       // (B,T,D)
  float* attn_ptr = out_ptr + (size_t)BT_ * D_;          // (B,H,T,T)

  // workspace layout
  char* ws = (char*)d_ws;
  size_t off = 0;
  bf16_t* posbuf = (bf16_t*)(ws + off); off += (size_t)B_ * H_ * T_ * T_ * sizeof(bf16_t); // 128 MB
  bf16_t* qu  = (bf16_t*)(ws + off); off += (size_t)BT_ * D_ * sizeof(bf16_t);
  bf16_t* qv  = (bf16_t*)(ws + off); off += (size_t)BT_ * D_ * sizeof(bf16_t);
  bf16_t* kbf = (bf16_t*)(ws + off); off += (size_t)BT_ * D_ * sizeof(bf16_t);
  bf16_t* vbf = (bf16_t*)(ws + off); off += (size_t)BT_ * D_ * sizeof(bf16_t);
  bf16_t* pbf = (bf16_t*)(ws + off); off += (size_t)T_  * D_ * sizeof(bf16_t);
  bf16_t* ctx = (bf16_t*)(ws + off); off += (size_t)BT_ * D_ * sizeof(bf16_t);

  const dim3 blk(256);

  // 1) projections
  {
    const int tiles = (BT_ / 16) * (D_ / 16);            // 8192
    const int grid = (tiles + WPB_ - 1) / WPB_;
    txl_proj_wmma<<<grid, blk, 0, stream>>>(query, Wq, bq, ub, vb, qu, qv, BT_);
    txl_proj_wmma<<<grid, blk, 0, stream>>>(key,   Wk, nullptr, nullptr, nullptr, kbf, nullptr, BT_);
    txl_proj_wmma<<<grid, blk, 0, stream>>>(value, Wv, nullptr, nullptr, nullptr, vbf, nullptr, BT_);
    const int ptiles = (T_ / 16) * (D_ / 16);            // 2048
    txl_proj_wmma<<<(ptiles + WPB_ - 1) / WPB_, blk, 0, stream>>>(
        encoding, Wpos, nullptr, nullptr, nullptr, pbf, nullptr, T_);
  }

  // 2) scores (TDM-staged): one block per 64x64 macro-tile
  {
    const int grid = B_ * H_ * (T_ / 64) * (T_ / 64);    // 16384
    txl_score_wmma<<<grid, blk, 0, stream>>>(qu, qv, kbf, pbf, attn_ptr, posbuf);
  }

  // 3) shift + mask + scale   4) softmax
  {
    const size_t nelem = (size_t)B_ * H_ * T_ * T_;
    txl_shift_mask_scale<<<(unsigned)(nelem / 2048), blk, 0, stream>>>(attn_ptr, posbuf, mask);
    txl_softmax<<<B_ * H_ * T_, blk, 0, stream>>>(attn_ptr);
  }

  // 5) context   6) output projection
  {
    const int ctiles = B_ * H_ * (T_ / 16);              // 2048 strips of 16x64
    txl_context_wmma<<<(ctiles + WPB_ - 1) / WPB_, blk, 0, stream>>>(attn_ptr, vbf, ctx);
    const int otiles = (BT_ / 16) * (D_ / 16);           // 8192
    txl_outproj_wmma<<<(otiles + WPB_ - 1) / WPB_, blk, 0, stream>>>(ctx, Wout, bout, out_ptr);
  }
}
